// ScatterAndGather_16690242912428
// MI455X (gfx1250) — matrix operand order repaired and checked by reading, not compiled, not run on hardware
//
#include <hip/hip_runtime.h>
#include <math.h>
#include <stdint.h>

typedef __attribute__((ext_vector_type(16))) __bf16 v16bf;
typedef __attribute__((ext_vector_type(8)))  __bf16 v8bf;
typedef __attribute__((ext_vector_type(8)))  float  v8f;
typedef unsigned short u16;

// round-to-nearest-even float -> bf16 (stored as u16; fragments reinterpret)
static __device__ __forceinline__ u16 f2bf(float f) {
  unsigned u = __float_as_uint(f);
  unsigned r = (u + 0x7FFFu + ((u >> 16) & 1u)) >> 16;
  return (u16)r;
}

// A fragment, contiguous K order: K = base .. base+15.
static __device__ __forceinline__ v16bf load_a_frag(const u16* ap) {
  v8bf lo = *(const v8bf*)(ap);       // K = base .. base+7
  v8bf hi = *(const v8bf*)(ap + 8);   // K = base+8 .. base+15
  v16bf a;
  #pragma unroll
  for (int i = 0; i < 8; ++i) { a[i] = lo[i]; a[i + 8] = hi[i]; }
  return a;
}

// ---------------- broadcast batched_x across T steps ----------------
__global__ void bcast_init(const float4* __restrict__ bx, float4* __restrict__ inp,
                           long perT4, long total4) {
  long i = (long)blockIdx.x * blockDim.x + threadIdx.x;
  if (i >= total4) return;
  inp[i] = bx[i % perT4];
}

// ---------------- scatter-add active nodes into full table ----------------
__global__ void scatter_add(const float* __restrict__ x, const int* __restrict__ idx,
                            float* __restrict__ inp, int node_num, long num_nodes,
                            int E, long total) {
  long gid = (long)blockIdx.x * blockDim.x + threadIdx.x;
  if (gid >= total) return;
  long row = gid / E;
  int  e   = (int)(gid - row * (long)E);
  int  t   = (int)(row / node_num);
  long dst = ((long)t * num_nodes + (long)idx[row]) * (long)E + e;
  atomicAdd(&inp[dst], x[gid]);
}

// ---------------- wave-per-row LayerNorm, fp32 in -> bf16 out ----------------
__global__ void ln_rows_bf16(const float* __restrict__ X, const float* __restrict__ sc,
                             const float* __restrict__ bi, u16* __restrict__ Y,
                             long rows, int D) {
  long row = (long)blockIdx.x * (blockDim.x >> 5) + (threadIdx.x >> 5);
  if (row >= rows) return;
  int lane = threadIdx.x & 31;
  const float* xr = X + row * (long)D;
  float s = 0.f, s2 = 0.f;
  for (int e = lane; e < D; e += 32) { float v = xr[e]; s += v; s2 += v * v; }
  #pragma unroll
  for (int off = 16; off > 0; off >>= 1) {
    s  += __shfl_xor(s,  off, 32);
    s2 += __shfl_xor(s2, off, 32);
  }
  float inv = 1.0f / (float)D;
  float mu  = s * inv;
  float var = s2 * inv - mu * mu;
  float rs  = rsqrtf(var + 1e-5f);
  u16* yr = Y + row * (long)D;
  for (int e = lane; e < D; e += 32)
    yr[e] = f2bf((xr[e] - mu) * rs * sc[e] + bi[e]);
}

// ---------------- gather (take_along_axis) + LayerNorm -> bf16 ----------------
__global__ void gather_ln_bf16(const float* __restrict__ H, const int* __restrict__ idx,
                               const float* __restrict__ sc, const float* __restrict__ bi,
                               u16* __restrict__ Y, long rows, int node_num,
                               long num_nodes, int D) {
  long row = (long)blockIdx.x * (blockDim.x >> 5) + (threadIdx.x >> 5);
  if (row >= rows) return;
  int lane = threadIdx.x & 31;
  int t = (int)(row / node_num);
  const float* xr = H + (((long)t * num_nodes) + (long)idx[row]) * (long)D;
  float s = 0.f, s2 = 0.f;
  for (int e = lane; e < D; e += 32) { float v = xr[e]; s += v; s2 += v * v; }
  #pragma unroll
  for (int off = 16; off > 0; off >>= 1) {
    s  += __shfl_xor(s,  off, 32);
    s2 += __shfl_xor(s2, off, 32);
  }
  float inv = 1.0f / (float)D;
  float mu  = s * inv;
  float var = s2 * inv - mu * mu;
  float rs  = rsqrtf(var + 1e-5f);
  u16* yr = Y + row * (long)D;
  for (int e = lane; e < D; e += 32)
    yr[e] = f2bf((xr[e] - mu) * rs * sc[e] + bi[e]);
}

// ---------------- pack fp32 weight [K,N] into WMMA B-fragment order (bf16) ----
// Bp[(kt*N + n)*32 + jj] = bf16(W[(kt*32 + jj)*N + n])
// so lane (n = lane&15, khalf = lane>>4) reads one contiguous v16bf per K-step.
__global__ void pack_b(const float* __restrict__ W, u16* __restrict__ Wp, int K, int N) {
  long gid = (long)blockIdx.x * blockDim.x + threadIdx.x;
  long total = (long)K * N;
  if (gid >= total) return;
  long per = (long)N * 32;
  long kt  = gid / per;
  long rem = gid - kt * per;
  int  n   = (int)(rem >> 5);
  int  jj  = (int)(rem & 31);
  Wp[gid] = f2bf(W[(kt * 32 + jj) * (long)N + n]);
}

// ---------------- WMMA GEMM: C[M,N] = bf16 A[M,K] x packed-bf16 B + bias -------
// One wave per 16x64 tile, 4 accumulators sharing one A fragment per K-step.
// Software-pipelined: fragments for K-step kt+1 are prefetched into fresh
// registers before the 4 WMMAs of step kt, so the per-WMMA waits are partial
// (loads of the next step stay in flight across the matrix ops).
template <bool GELU_BF16_OUT>
__global__ void wmma_gemm(const u16* __restrict__ A, const u16* __restrict__ Bp,
                          const float* __restrict__ bias, void* __restrict__ Out,
                          int M, int N, int K) {
  int  wave = threadIdx.x >> 5;
  int  lane = threadIdx.x & 31;
  long tile = (long)blockIdx.x * (blockDim.x >> 5) + wave;
  int  tilesN = N >> 6;                       // 64 columns per wave tile
  long totalTiles = (long)(M >> 4) * tilesN;
  if (tile >= totalTiles) return;             // wave-uniform: EXEC all-1s for WMMA
  int tm = (int)(tile / tilesN);
  int tn = (int)(tile % tilesN);

  int half = lane >> 4;                       // K-half selector
  int mrow = lane & 15;                       // A row within tile / B column
  int base = half << 4;                       // 0 or 16

  const u16* ap    = A + (size_t)(tm * 16 + mrow) * K + base;
  const u16* bp    = Bp + (((size_t)(tn * 64 + mrow)) << 5) + (half << 4);
  size_t     bstep = (size_t)N << 5;          // advance one K-tile in packed B

  v8f acc[4] = {v8f{}, v8f{}, v8f{}, v8f{}};
  int ktEnd = K >> 5;                         // >= 4 for all our GEMMs

  // prologue: fragments for K-step 0
  v16bf a  = load_a_frag(ap);
  v16bf b0 = *(const v16bf*)(bp);
  v16bf b1 = *(const v16bf*)(bp + 512);       // 16 cols * 32 elems
  v16bf b2 = *(const v16bf*)(bp + 1024);
  v16bf b3 = *(const v16bf*)(bp + 1536);

  for (int kt = 0; kt < ktEnd - 1; ++kt) {
    ap += 32;
    bp += bstep;
    // prefetch next K-step into fresh registers (keeps loads in flight
    // across the WMMAs below -> partial s_wait_loadcnt, not 0)
    v16bf an  = load_a_frag(ap);
    v16bf bn0 = *(const v16bf*)(bp);
    v16bf bn1 = *(const v16bf*)(bp + 512);
    v16bf bn2 = *(const v16bf*)(bp + 1024);
    v16bf bn3 = *(const v16bf*)(bp + 1536);

    acc[0] = __builtin_amdgcn_wmma_f32_16x16x32_bf16(false, a, false, b0,
                                                     (short)0, acc[0], false, false);
    acc[1] = __builtin_amdgcn_wmma_f32_16x16x32_bf16(false, a, false, b1,
                                                     (short)0, acc[1], false, false);
    acc[2] = __builtin_amdgcn_wmma_f32_16x16x32_bf16(false, a, false, b2,
                                                     (short)0, acc[2], false, false);
    acc[3] = __builtin_amdgcn_wmma_f32_16x16x32_bf16(false, a, false, b3,
                                                     (short)0, acc[3], false, false);
    a = an; b0 = bn0; b1 = bn1; b2 = bn2; b3 = bn3;
  }
  // epilogue K-step
  acc[0] = __builtin_amdgcn_wmma_f32_16x16x32_bf16(false, a, false, b0,
                                                   (short)0, acc[0], false, false);
  acc[1] = __builtin_amdgcn_wmma_f32_16x16x32_bf16(false, a, false, b1,
                                                   (short)0, acc[1], false, false);
  acc[2] = __builtin_amdgcn_wmma_f32_16x16x32_bf16(false, a, false, b2,
                                                   (short)0, acc[2], false, false);
  acc[3] = __builtin_amdgcn_wmma_f32_16x16x32_bf16(false, a, false, b3,
                                                   (short)0, acc[3], false, false);

  int row0 = tm * 16 + half * 8;              // C layout: VGPR r -> M = r + 8*(lane>>4)
  #pragma unroll
  for (int s = 0; s < 4; ++s) {
    int   col = tn * 64 + s * 16 + mrow;
    float bv  = bias[col];
    if (GELU_BF16_OUT) {
      u16* O = (u16*)Out;
      #pragma unroll
      for (int r = 0; r < 8; ++r) {
        float v = acc[s][r] + bv;
        float g = 0.5f * v * (1.0f + erff(v * 0.70710678118654752f)); // exact GELU
        O[(size_t)(row0 + r) * N + col] = f2bf(g);
      }
    } else {
      float* O = (float*)Out;
      #pragma unroll
      for (int r = 0; r < 8; ++r)
        O[(size_t)(row0 + r) * N + col] = acc[s][r] + bv;
    }
  }
}

extern "C" void kernel_launch(void* const* d_in, const int* in_sizes, int n_in,
                              void* d_out, int out_size, void* d_ws, size_t ws_size,
                              hipStream_t stream) {
  const float* x      = (const float*)d_in[0];
  const float* bx     = (const float*)d_in[1];
  const int*   idx    = (const int*)d_in[2];
  const float* ln_d_s = (const float*)d_in[3];
  const float* ln_d_b = (const float*)d_in[4];
  const float* W1d    = (const float*)d_in[5];
  const float* b1d    = (const float*)d_in[6];
  const float* W2d    = (const float*)d_in[7];
  const float* b2d    = (const float*)d_in[8];
  const float* ln_u_s = (const float*)d_in[9];
  const float* ln_u_b = (const float*)d_in[10];
  const float* W1u    = (const float*)d_in[11];
  const float* b1u    = (const float*)d_in[12];
  const float* W2u    = (const float*)d_in[13];
  const float* b2u    = (const float*)d_in[14];

  const int  E   = in_sizes[3];              // 256
  const int  C   = in_sizes[9];              // 128
  const long TN  = in_sizes[2];              // T*node_num = 40000
  const int  T   = 4;                        // reference setup
  const int  node_num = (int)(TN / T);       // 10000
  const long num_nodes = (long)in_sizes[1] / E;  // 50000
  const int  N1d = in_sizes[6];              // 2E = 512
  const int  N1u = in_sizes[12];             // 2C = 256
  const int  N2u = in_sizes[14];             // E  = 256
  const long TN2 = (long)T * num_nodes;      // 200000

  // ---- workspace carving (aliased: inp<->act1, a_d<->h) ----
  uint8_t* ws = (uint8_t*)d_ws;
  size_t off = 0;
  auto carve = [&](size_t bytes) -> size_t {
    size_t o = off; off = (off + bytes + 255) & ~(size_t)255; return o;
  };
  size_t szInp  = (size_t)TN2 * E   * 4;     // inp fp32        (204.8 MB)
  size_t szAct1 = (size_t)TN2 * N1d * 2;     // act1 bf16       (204.8 MB)
  size_t oA  = carve(szInp > szAct1 ? szInp : szAct1);
  size_t szAd = (size_t)TN2 * E * 2;         // a_d bf16        (102.4 MB)
  size_t szH  = (size_t)TN2 * C * 4;         // h fp32          (102.4 MB)
  size_t oB  = carve(szAd > szH ? szAd : szH);
  size_t oG   = carve((size_t)TN * C   * 2); // g bf16
  size_t oAc2 = carve((size_t)TN * N1u * 2); // act2 bf16
  size_t oW1d = carve((size_t)E   * N1d * 2);
  size_t oW2d = carve((size_t)N1d * C   * 2);
  size_t oW1u = carve((size_t)C   * N1u * 2);
  size_t oW2u = carve((size_t)N1u * N2u * 2);
  (void)ws_size; (void)n_in;

  float* inp  = (float*)(ws + oA);
  u16*   act1 = (u16*)  (ws + oA);
  u16*   a_d  = (u16*)  (ws + oB);
  float* h    = (float*)(ws + oB);
  u16*   g    = (u16*)  (ws + oG);
  u16*   act2 = (u16*)  (ws + oAc2);
  u16*   Wp1d = (u16*)(ws + oW1d);
  u16*   Wp2d = (u16*)(ws + oW2d);
  u16*   Wp1u = (u16*)(ws + oW1u);
  u16*   Wp2u = (u16*)(ws + oW2u);

  // ---- pack weights into WMMA B-fragment order (tiny, stays in L2) ----
  {
    long n;
    n = (long)E * N1d;   pack_b<<<(unsigned)((n + 255) / 256), 256, 0, stream>>>(W1d, Wp1d, E,   N1d);
    n = (long)N1d * C;   pack_b<<<(unsigned)((n + 255) / 256), 256, 0, stream>>>(W2d, Wp2d, N1d, C);
    n = (long)C * N1u;   pack_b<<<(unsigned)((n + 255) / 256), 256, 0, stream>>>(W1u, Wp1u, C,   N1u);
    n = (long)N1u * N2u; pack_b<<<(unsigned)((n + 255) / 256), 256, 0, stream>>>(W2u, Wp2u, N1u, N2u);
  }

  // ---- inp = broadcast(batched_x) ; scatter-add x ----
  {
    long perT4  = num_nodes * (long)E / 4;
    long total4 = TN2 * (long)E / 4;
    bcast_init<<<(unsigned)((total4 + 255) / 256), 256, 0, stream>>>(
        (const float4*)bx, (float4*)inp, perT4, total4);
    long total = TN * (long)E;
    scatter_add<<<(unsigned)((total + 255) / 256), 256, 0, stream>>>(
        x, idx, inp, node_num, num_nodes, E, total);
  }

  // ---- LN(E) -> a_d bf16 ----
  ln_rows_bf16<<<(unsigned)((TN2 + 3) / 4), 128, 0, stream>>>(
      inp, ln_d_s, ln_d_b, a_d, TN2, E);

  // ---- down MLP: GEMM1 (+GELU, bf16) then GEMM2 (fp32 h) ----
  {
    long tiles = (TN2 >> 4) * (long)(N1d >> 6);
    wmma_gemm<true><<<(unsigned)((tiles + 3) / 4), 128, 0, stream>>>(
        a_d, Wp1d, b1d, (void*)act1, (int)TN2, N1d, E);
    tiles = (TN2 >> 4) * (long)(C >> 6);
    wmma_gemm<false><<<(unsigned)((tiles + 3) / 4), 128, 0, stream>>>(
        act1, Wp2d, b2d, (void*)h, (int)TN2, C, N1d);
  }

  // ---- gather + LN(C) -> g bf16 ----
  gather_ln_bf16<<<(unsigned)((TN + 3) / 4), 128, 0, stream>>>(
      h, idx, ln_u_s, ln_u_b, g, TN, node_num, num_nodes, C);

  // ---- up MLP: GEMM1 (+GELU, bf16) then GEMM2 (fp32 -> d_out) ----
  {
    long tiles = (TN >> 4) * (long)(N1u >> 6);
    wmma_gemm<true><<<(unsigned)((tiles + 3) / 4), 128, 0, stream>>>(
        g, Wp1u, b1u, (void*)act2, (int)TN, N1u, C);
    tiles = (TN >> 4) * (long)(N2u >> 6);
    wmma_gemm<false><<<(unsigned)((tiles + 3) / 4), 128, 0, stream>>>(
        act2, Wp2u, b2u, d_out, (int)TN, N2u, N1u);
  }
  (void)out_size;
}
